// MultiHeaderModel_1898375544934
// MI455X (gfx1250) — matrix operand order repaired and checked by reading, not compiled, not run on hardware
//
#include <hip/hip_runtime.h>
#include <hip/hip_bf16.h>
#include <math.h>

typedef __attribute__((ext_vector_type(16))) _Float16 v16h;
typedef __attribute__((ext_vector_type(8)))  _Float16 v8h;
typedef __attribute__((ext_vector_type(8)))  float    v8f;
typedef __attribute__((ext_vector_type(4)))  float    v4f;

#define BB 8
#define LL 256
#define ROWS (BB*LL)          // 2048
#define DIN 256               // 2*CHAR_EMBED
#define UNITS 64
#define HSENT 128             // 2*UNITS
#define ENTN 16
#define ENTE 64
#define RELE 128
#define RELN 24
#define HREL 192              // HSENT + ENTE

__device__ __forceinline__ float sigmf(float x) { return 1.0f / (1.0f + expf(-x)); }

// ---------------- WMMA fragment helpers (ISA 7.12.2 layouts) ----------------
// A: row-major M x K (f16). 16x32 tile fragment: lanes 0-15 K=ko..ko+7 / ko+16..23 with ko=0,
// lanes 16-31 same pattern with ko=8. Both 8-half groups are 16B-aligned -> b128 loads.
__device__ __forceinline__ v16h load_a_frag(const _Float16* __restrict__ A, int lda,
                                            int row0, int k0, int lane) {
  int m  = row0 + (lane & 15);
  int ko = (lane < 16) ? 0 : 8;
  const _Float16* p = A + (size_t)m * lda + k0 + ko;
  v8h lo = *(const v8h*)(p);        // K = ko+0..7
  v8h hi = *(const v8h*)(p + 16);   // K = ko+16..23
  v16h a;
#pragma unroll
  for (int i = 0; i < 8; ++i) { a[i] = lo[i]; a[8 + i] = hi[i]; }
  return a;
}
// B stored TRANSPOSED: BT is N x K row-major. Fragment: lane covers column n = col0+(lane&15),
// halves i=0..7 are K = k0+ko+i, i=8..15 are K = k0+ko+16+(i-8), with ko = 0 (lanes 0-15) / 8
// (lanes 16-31) -> two b128 loads.
__device__ __forceinline__ v16h load_bT_frag(const _Float16* __restrict__ BT, int ldbt,
                                             int k0, int col0, int lane) {
  int n  = col0 + (lane & 15);
  int ko = (lane < 16) ? 0 : 8;
  const _Float16* p = BT + (size_t)n * ldbt + k0 + ko;
  v8h lo = *(const v8h*)(p);
  v8h hi = *(const v8h*)(p + 16);
  v16h b;
#pragma unroll
  for (int i = 0; i < 8; ++i) { b[i] = lo[i]; b[8 + i] = hi[i]; }
  return b;
}

// C(MxN,f32) = A(MxK,f16) @ BT(NxK,f16)^T + bias[N], optional ReLU.
// One wave per 16x16 C tile. grid=(N/16, M/16), block=32. K must be a multiple of 64.
__global__ void wmma_gemm_kernel(const _Float16* __restrict__ A,
                                 const _Float16* __restrict__ BT,
                                 const float* __restrict__ bias,
                                 float* __restrict__ C,
                                 int M, int N, int K, int relu) {
  int lane = threadIdx.x & 31;
  int col0 = blockIdx.x * 16;
  int row0 = blockIdx.y * 16;
  v8f acc = {};
  for (int k0 = 0; k0 < K; k0 += 64) {
    // prefetch next K-block (speculative; dropped if OOB)
    __builtin_prefetch(A + (size_t)(row0 + (lane & 15)) * K + k0 + 64, 0, 3);
    __builtin_prefetch(BT + (size_t)(col0 + (lane & 15)) * K + k0 + 64, 0, 3);
    v16h a0 = load_a_frag(A, K, row0, k0, lane);
    v16h b0 = load_bT_frag(BT, K, k0, col0, lane);
    acc = __builtin_amdgcn_wmma_f32_16x16x32_f16(false, a0, false, b0, (short)0, acc,
                                                 false, false);
    v16h a1 = load_a_frag(A, K, row0, k0 + 32, lane);
    v16h b1 = load_bT_frag(BT, K, k0 + 32, col0, lane);
    acc = __builtin_amdgcn_wmma_f32_16x16x32_f16(false, a1, false, b1, (short)0, acc,
                                                 false, false);
  }
  int n = col0 + (lane & 15);
  float bs = bias ? bias[n] : 0.0f;
#pragma unroll
  for (int r = 0; r < 8; ++r) {
    int m = row0 + ((lane < 16) ? r : r + 8);
    float v = acc[r] + bs;
    if (relu) v = fmaxf(v, 0.0f);
    C[(size_t)m * N + n] = v;
  }
}

// ---------------- misc kernels ----------------
__global__ void f32_to_f16_kernel(const float* __restrict__ in, _Float16* __restrict__ out, int n) {
  int i = blockIdx.x * blockDim.x + threadIdx.x;
  if (i < n) out[i] = (_Float16)in[i];
}

// in: K x N (f32) -> out: N x K (f16)   (weight transpose for WMMA B operand)
__global__ void f32_to_f16T_kernel(const float* __restrict__ in, _Float16* __restrict__ out,
                                   int K, int N) {
  int idx = blockIdx.x * blockDim.x + threadIdx.x;
  if (idx >= K * N) return;
  int k = idx / N, n = idx % N;
  out[(size_t)n * K + k] = (_Float16)in[idx];
}

// embed_h[row, col] = col<128 ? char_table[cid][col] : word_table[wid][col-128]
__global__ void gather_embed_kernel(const int* __restrict__ char_ids, const int* __restrict__ word_ids,
                                    const float* __restrict__ char_table, const float* __restrict__ word_table,
                                    _Float16* __restrict__ embed_h) {
  int row = blockIdx.x, col = threadIdx.x;
  float v = (col < 128) ? char_table[(size_t)char_ids[row] * 128 + col]
                        : word_table[(size_t)word_ids[row] * 128 + (col - 128)];
  embed_h[(size_t)row * DIN + col] = (_Float16)v;
}

__global__ void mask_kernel(const int* __restrict__ char_ids, int* __restrict__ mask, int* __restrict__ tlens) {
  int i = blockIdx.x * blockDim.x + threadIdx.x;
  if (i < ROWS) mask[i] = (char_ids[i] != 0) ? 1 : 0;
  if (i < BB) {
    int c = 0;
    for (int t = 0; t < LL; ++t) c += (char_ids[i * LL + t] != 0) ? 1 : 0;
    tlens[i] = c;
  }
}

// One workgroup per direction. h,c,z in static LDS; rkernel (64KB) in dynamic LDS
// (legal on CDNA5: up to 320KB LDS per workgroup). 256 sequential steps.
__global__ void lstm_kernel(const float* __restrict__ xz_f, const float* __restrict__ xz_b,
                            const float* __restrict__ rk_f, const float* __restrict__ rk_b,
                            const int* __restrict__ mask, float* __restrict__ sent) {
  int dir = blockIdx.x;
  const float* xz = dir ? xz_b : xz_f;
  const float* rk = dir ? rk_b : rk_f;   // (64, 256)
  __shared__ float h[BB][UNITS];
  __shared__ float c[BB][UNITS];
  __shared__ float z[BB][4 * UNITS];
  extern __shared__ float rks[];         // 64*256 floats = 64KB
  int tid = threadIdx.x; // 256 threads
  for (int i = tid; i < UNITS * 256; i += 256) rks[i] = rk[i];
  for (int i = tid; i < BB * UNITS; i += 256) { ((float*)h)[i] = 0.0f; ((float*)c)[i] = 0.0f; }
  __syncthreads();
  for (int s = 0; s < LL; ++s) {
    int t = dir ? (LL - 1 - s) : s;
    // z[b][n] = xz[b*L+t][n] + sum_k h[b][k]*rks[k][n]
    for (int idx = tid; idx < BB * 256; idx += 256) {
      int b = idx >> 8, n = idx & 255;
      float acc = xz[((size_t)(b * LL + t)) * 256 + n];
#pragma unroll 8
      for (int k = 0; k < UNITS; ++k) acc += h[b][k] * rks[k * 256 + n];
      z[b][n] = acc;
    }
    __syncthreads();
    for (int idx = tid; idx < BB * UNITS; idx += 256) {
      int b = idx >> 6, u = idx & 63;
      float zi = z[b][u], zf = z[b][64 + u], zg = z[b][128 + u], zo = z[b][192 + u];
      float cn = sigmf(zf) * c[b][u] + sigmf(zi) * tanhf(zg);
      float hn = sigmf(zo) * tanhf(cn);
      if (mask[b * LL + t]) { c[b][u] = cn; h[b][u] = hn; }
      sent[((size_t)(b * LL + t)) * HSENT + dir * UNITS + u] = h[b][u];
    }
    __syncthreads();
  }
}

__global__ void softmax16_kernel(const float* __restrict__ in, float* __restrict__ out) {
  int row = blockIdx.x * blockDim.x + threadIdx.x;
  if (row >= ROWS) return;
  const float* p = in + (size_t)row * ENTN;
  float mx = p[0];
#pragma unroll
  for (int k = 1; k < ENTN; ++k) mx = fmaxf(mx, p[k]);
  float e[ENTN], s = 0.0f;
#pragma unroll
  for (int k = 0; k < ENTN; ++k) { e[k] = expf(p[k] - mx); s += e[k]; }
  float inv = 1.0f / s;
#pragma unroll
  for (int k = 0; k < ENTN; ++k) out[(size_t)row * ENTN + k] = e[k] * inv;
}

// One lane per batch. scores = ent_logits (post-softmax), trans (16x16).
__global__ void viterbi_kernel(const float* __restrict__ scores, const float* __restrict__ trans,
                               const int* __restrict__ tlens, int* __restrict__ tags,
                               unsigned char* __restrict__ bp) {
  int b = threadIdx.x;
  if (b >= BB) return;
  const float* sc = scores + (size_t)b * LL * ENTN;
  float alpha[ENTN], na[ENTN];
#pragma unroll
  for (int e = 0; e < ENTN; ++e) alpha[e] = sc[e];
  int tl = tlens[b];
  for (int t = 1; t < LL; ++t) {
    bool valid = (t < tl);
    for (int e = 0; e < ENTN; ++e) {
      float best = -3.0e38f; int bi = 0;
      for (int p = 0; p < ENTN; ++p) {
        float v = alpha[p] + trans[p * ENTN + e];
        if (v > best) { best = v; bi = p; }
      }
      na[e] = best + sc[(size_t)t * ENTN + e];
      bp[((size_t)b * (LL - 1) + (t - 1)) * ENTN + e] = valid ? (unsigned char)bi : (unsigned char)e;
    }
    if (valid) {
#pragma unroll
      for (int e = 0; e < ENTN; ++e) alpha[e] = na[e];
    }
  }
  int last = 0; float bb = alpha[0];
#pragma unroll
  for (int e = 1; e < ENTN; ++e) if (alpha[e] > bb) { bb = alpha[e]; last = e; }
  int cur = last;
  tags[b * LL + (LL - 1)] = last;
  for (int t = LL - 1; t >= 1; --t) {
    cur = bp[((size_t)b * (LL - 1) + (t - 1)) * ENTN + cur];
    tags[b * LL + (t - 1)] = cur;
  }
  for (int t = 0; t < LL; ++t) if (t >= tl) tags[b * LL + t] = 0;
}

__global__ void build_relenc_kernel(const float* __restrict__ sent, const float* __restrict__ ent_table,
                                    const int* __restrict__ tags, _Float16* __restrict__ relh) {
  int idx = blockIdx.x * blockDim.x + threadIdx.x;
  if (idx >= ROWS * HREL) return;
  int row = idx / HREL, col = idx % HREL;
  float v = (col < HSENT) ? sent[(size_t)row * HSENT + col]
                          : ent_table[(size_t)tags[row] * ENTE + (col - HSENT)];
  relh[idx] = (_Float16)v;
}

// Mu[d][k] = sum_e W_uv[d][e]*W_rel[e][k]; Mv uses rows 128..255; c[k]=b_uv@W_rel+b_rel
__global__ void build_M_kernel(const float* __restrict__ W_uv, const float* __restrict__ W_rel,
                               const float* __restrict__ b_uv, const float* __restrict__ b_rel,
                               float* __restrict__ Mu, float* __restrict__ Mv, float* __restrict__ cvec) {
  int idx = blockIdx.x * blockDim.x + threadIdx.x;
  if (idx < RELE * RELN) {
    int d = idx / RELN, k = idx % RELN;
    float su = 0.0f, sv = 0.0f;
    for (int e = 0; e < RELE; ++e) {
      float wr = W_rel[e * RELN + k];
      su += W_uv[(size_t)d * RELE + e] * wr;
      sv += W_uv[(size_t)(RELE + d) * RELE + e] * wr;
    }
    Mu[idx] = su; Mv[idx] = sv;
  }
  if (idx < RELN) {
    float s = b_rel[idx];
    for (int e = 0; e < RELE; ++e) s += b_uv[e] * W_rel[e * RELN + idx];
    cvec[idx] = s;
  }
}

// Q[row][k] = sum_d U[row][d] * Mw[d][k]   (2048x128 @ 128x24)
__global__ void qproj_kernel(const float* __restrict__ U, const float* __restrict__ Mw,
                             float* __restrict__ Q) {
  int idx = blockIdx.x * blockDim.x + threadIdx.x;
  if (idx >= ROWS * RELN) return;
  int row = idx / RELN, k = idx % RELN;
  const float* up = U + (size_t)row * RELE;
  float s = 0.0f;
#pragma unroll 8
  for (int d = 0; d < RELE; ++d) s += up[d] * Mw[d * RELN + k];
  Q[idx] = s;
}

// out[b,i,j,:] = softmax(Qu[b,j,:] + Qv[b,i,:] + c) — 50 MB streaming write, pure b128 traffic.
__global__ void rel_out_kernel(const float* __restrict__ Qu, const float* __restrict__ Qv,
                               const float* __restrict__ cvec, float* __restrict__ out) {
  int idx = blockIdx.x * blockDim.x + threadIdx.x;
  if (idx >= BB * LL * LL) return;
  int j = idx & 255, i = (idx >> 8) & 255, b = idx >> 16;
  const v4f* qu4 = (const v4f*)(Qu + (size_t)(b * LL + j) * RELN);   // 96B-aligned rows
  const v4f* qv4 = (const v4f*)(Qv + (size_t)(b * LL + i) * RELN);
  const v4f* cc4 = (const v4f*)cvec;
  float vals[RELN]; float mx = -3.0e38f;
#pragma unroll
  for (int q = 0; q < RELN / 4; ++q) {
    v4f a = qu4[q], bq = qv4[q], cq = cc4[q];
#pragma unroll
    for (int e = 0; e < 4; ++e) {
      float v = a[e] + bq[e] + cq[e];
      vals[q * 4 + e] = v; mx = fmaxf(mx, v);
    }
  }
  float s = 0.0f;
#pragma unroll
  for (int k = 0; k < RELN; ++k) { vals[k] = expf(vals[k] - mx); s += vals[k]; }
  float inv = 1.0f / s;
  v4f* o4 = (v4f*)(out + (size_t)idx * RELN);
#pragma unroll
  for (int q = 0; q < RELN / 4; ++q) {
    v4f t;
#pragma unroll
    for (int e = 0; e < 4; ++e) t[e] = vals[q * 4 + e] * inv;
    o4[q] = t;
  }
}

__global__ void mask_out_kernel(const int* __restrict__ mask, float* __restrict__ out) {
  int i = blockIdx.x * blockDim.x + threadIdx.x;
  if (i < ROWS) out[i] = mask[i] ? 1.0f : 0.0f;
}

// ---------------- host ----------------
extern "C" void kernel_launch(void* const* d_in, const int* in_sizes, int n_in,
                              void* d_out, int out_size, void* d_ws, size_t ws_size,
                              hipStream_t stream) {
  (void)in_sizes; (void)n_in; (void)out_size; (void)ws_size;
  const int*   char_ids  = (const int*)  d_in[0];
  const int*   word_ids  = (const int*)  d_in[1];
  const float* char_tab  = (const float*)d_in[2];
  const float* word_tab  = (const float*)d_in[3];
  const float* ent_tab   = (const float*)d_in[4];
  const float* trans     = (const float*)d_in[5];
  const float* k_f       = (const float*)d_in[6];
  const float* rk_f      = (const float*)d_in[7];
  const float* b_f       = (const float*)d_in[8];
  const float* k_b       = (const float*)d_in[9];
  const float* rk_b      = (const float*)d_in[10];
  const float* b_b       = (const float*)d_in[11];
  const float* W_ent     = (const float*)d_in[12];
  const float* b_ent     = (const float*)d_in[13];
  const float* W_u       = (const float*)d_in[14];
  const float* b_u       = (const float*)d_in[15];
  const float* W_v       = (const float*)d_in[16];
  const float* b_v       = (const float*)d_in[17];
  const float* W_uv      = (const float*)d_in[18];
  const float* b_uv      = (const float*)d_in[19];
  const float* W_rel     = (const float*)d_in[20];
  const float* b_rel     = (const float*)d_in[21];

  char* ws = (char*)d_ws;
  size_t off = 0;
  auto alloc = [&](size_t bytes) -> char* {
    char* p = ws + off;
    off = (off + bytes + 255) & ~(size_t)255;
    return p;
  };
  _Float16* embed_h = (_Float16*)alloc((size_t)ROWS * DIN * 2);
  _Float16* kfhT    = (_Float16*)alloc((size_t)DIN * 256 * 2);   // 256 x 256 (N x K)
  _Float16* kbhT    = (_Float16*)alloc((size_t)DIN * 256 * 2);
  _Float16* wenthT  = (_Float16*)alloc((size_t)ENTN * HSENT * 2); // 16 x 128
  _Float16* wuhT    = (_Float16*)alloc((size_t)RELE * HREL * 2);  // 128 x 192
  _Float16* wvhT    = (_Float16*)alloc((size_t)RELE * HREL * 2);
  float*    xzf     = (float*)   alloc((size_t)ROWS * 256 * 4);
  float*    xzb     = (float*)   alloc((size_t)ROWS * 256 * 4);
  int*      maski   = (int*)     alloc((size_t)ROWS * 4);
  int*      tlens   = (int*)     alloc((size_t)BB * 4);
  float*    sent    = (float*)   alloc((size_t)ROWS * HSENT * 4);
  _Float16* senth   = (_Float16*)alloc((size_t)ROWS * HSENT * 2);
  float*    entpre  = (float*)   alloc((size_t)ROWS * ENTN * 4);
  int*      tags    = (int*)     alloc((size_t)ROWS * 4);
  unsigned char* bp = (unsigned char*)alloc((size_t)BB * (LL - 1) * ENTN);
  _Float16* relh    = (_Float16*)alloc((size_t)ROWS * HREL * 2);
  float*    ubuf    = (float*)   alloc((size_t)ROWS * RELE * 4);
  float*    vbuf    = (float*)   alloc((size_t)ROWS * RELE * 4);
  float*    Mu      = (float*)   alloc((size_t)RELE * RELN * 4);
  float*    Mv      = (float*)   alloc((size_t)RELE * RELN * 4);
  float*    cvec    = (float*)   alloc(32 * 4);
  float*    Qu      = (float*)   alloc((size_t)ROWS * RELN * 4);
  float*    Qv      = (float*)   alloc((size_t)ROWS * RELN * 4);

  float* out_ent  = (float*)d_out;                               // 2048*16
  float* out_rel  = out_ent + (size_t)ROWS * ENTN;               // 8*256*256*24
  float* out_mask = out_rel + (size_t)BB * LL * LL * RELN;       // 2048

  // 1) embeddings + mask
  gather_embed_kernel<<<ROWS, DIN, 0, stream>>>(char_ids, word_ids, char_tab, word_tab, embed_h);
  mask_kernel<<<(ROWS + 255) / 256, 256, 0, stream>>>(char_ids, maski, tlens);

  // 2) f16 weight conversions, transposed for the WMMA B operand
  f32_to_f16T_kernel<<<(DIN * 256 + 255) / 256, 256, 0, stream>>>(k_f, kfhT, DIN, 256);
  f32_to_f16T_kernel<<<(DIN * 256 + 255) / 256, 256, 0, stream>>>(k_b, kbhT, DIN, 256);
  f32_to_f16T_kernel<<<(HSENT * ENTN + 255) / 256, 256, 0, stream>>>(W_ent, wenthT, HSENT, ENTN);
  f32_to_f16T_kernel<<<(HREL * RELE + 255) / 256, 256, 0, stream>>>(W_u, wuhT, HREL, RELE);
  f32_to_f16T_kernel<<<(HREL * RELE + 255) / 256, 256, 0, stream>>>(W_v, wvhT, HREL, RELE);

  // 3) xz = embed @ kernel + bias  (WMMA, M=2048 N=256 K=256)
  wmma_gemm_kernel<<<dim3(256 / 16, ROWS / 16), 32, 0, stream>>>(embed_h, kfhT, b_f, xzf, ROWS, 256, DIN, 0);
  wmma_gemm_kernel<<<dim3(256 / 16, ROWS / 16), 32, 0, stream>>>(embed_h, kbhT, b_b, xzb, ROWS, 256, DIN, 0);

  // 4) BiLSTM recurrence (2 workgroups: fwd, bwd; 64KB dynamic LDS for rkernel)
  lstm_kernel<<<2, 256, UNITS * 256 * sizeof(float), stream>>>(xzf, xzb, rk_f, rk_b, maski, sent);

  // 5) ent head: sent @ W_ent + b_ent (WMMA), softmax16 -> d_out
  f32_to_f16_kernel<<<(ROWS * HSENT + 255) / 256, 256, 0, stream>>>(sent, senth, ROWS * HSENT);
  wmma_gemm_kernel<<<dim3(ENTN / 16, ROWS / 16), 32, 0, stream>>>(senth, wenthT, b_ent, entpre, ROWS, ENTN, HSENT, 0);
  softmax16_kernel<<<(ROWS + 255) / 256, 256, 0, stream>>>(entpre, out_ent);

  // 6) Viterbi on softmaxed scores
  viterbi_kernel<<<1, 32, 0, stream>>>(out_ent, trans, tlens, tags, bp);

  // 7) rel_enc = [sent, ent_table[tags]] (f16); u/v = relu(rel_enc @ W + b) (WMMA)
  build_relenc_kernel<<<(ROWS * HREL + 255) / 256, 256, 0, stream>>>(sent, ent_tab, tags, relh);
  wmma_gemm_kernel<<<dim3(RELE / 16, ROWS / 16), 32, 0, stream>>>(relh, wuhT, b_u, ubuf, ROWS, RELE, HREL, 1);
  wmma_gemm_kernel<<<dim3(RELE / 16, ROWS / 16), 32, 0, stream>>>(relh, wvhT, b_v, vbuf, ROWS, RELE, HREL, 1);

  // 8) rank decomposition: Mu/Mv/c, then Qu = u@Mu, Qv = v@Mv
  build_M_kernel<<<(RELE * RELN + 255) / 256, 256, 0, stream>>>(W_uv, W_rel, b_uv, b_rel, Mu, Mv, cvec);
  qproj_kernel<<<(ROWS * RELN + 255) / 256, 256, 0, stream>>>(ubuf, Mu, Qu);
  qproj_kernel<<<(ROWS * RELN + 255) / 256, 256, 0, stream>>>(vbuf, Mv, Qv);

  // 9) streaming output: softmax24(Qu[b,j]+Qv[b,i]+c) -> 50 MB write (BW-bound)
  rel_out_kernel<<<(BB * LL * LL + 255) / 256, 256, 0, stream>>>(Qu, Qv, cvec, out_rel);
  mask_out_kernel<<<(ROWS + 255) / 256, 256, 0, stream>>>(maski, out_mask);
}
